// MultiheadAttention_24721831756165
// MI455X (gfx1250) — compile-verified, matrix operand order checked by reading, not run on hardware
//
#include <hip/hip_runtime.h>

// ---------------------------------------------------------------------------
// MultiheadAttention forward for gfx1250 (MI455X).
// f16 WMMA everywhere (v_wmma_f32_16x16x32_f16), flash-style fused attention.
// GEMM operand staging uses the CDNA5 Tensor Data Mover (tensor_load_to_lds,
// TENSORcnt); attention staging uses async global->LDS copies (ASYNCcnt).
// Both are double-buffered so stage k+1 streams into LDS during stage k math.
// ---------------------------------------------------------------------------

typedef _Float16 h8   __attribute__((ext_vector_type(8)));
typedef _Float16 h16  __attribute__((ext_vector_type(16)));
typedef float    f8   __attribute__((ext_vector_type(8)));
typedef uint32_t u32x4 __attribute__((ext_vector_type(4)));
typedef uint32_t u32x8 __attribute__((ext_vector_type(8)));

#define B_   4
#define S_   2048
#define E_   1024
#define H_   16
#define D_   64
#define K3_  3072
#define M_   8192   // B_*S_

// CDNA5 async copy: 16B global -> LDS, tracked on ASYNCcnt (no VGPR staging).
__device__ __forceinline__ void async_cp_b128(uint32_t lds_off,
                                              const _Float16* g) {
  asm volatile("global_load_async_to_lds_b128 %0, %1, off"
               :: "v"(lds_off), "v"((unsigned long long)(uintptr_t)g)
               : "memory");
}
// Truncate generic pointer to LDS byte address (aperture: addr[31:0] = offset).
__device__ __forceinline__ uint32_t lds_off(const void* p) {
  return (uint32_t)(uintptr_t)p;
}
#define ASYNC_WAIT(n) asm volatile("s_wait_asynccnt " #n ::: "memory")

// Tensor Data Mover: DMA a 2-D tile (128 rows x 64 f16) from a row-major
// [rows x Kelems] f16 tensor into LDS (packed [128][64]). D# per ISA
// ch8.3/8.4; 2-D tile -> descriptor groups 2/3 omitted (NULL).
__device__ __forceinline__ void tdm_load_tile64(const _Float16* tile_start,
                                                uint32_t ldsaddr,
                                                uint32_t Kelems,
                                                uint32_t rows) {
  const uint64_t ga = (uint64_t)(uintptr_t)tile_start;
  u32x4 g0;
  g0[0] = 1u;                                            // count=1 (valid D#)
  g0[1] = ldsaddr;                                       // lds_addr (bytes)
  g0[2] = (uint32_t)ga;                                  // global_addr[31:0]
  g0[3] = (uint32_t)((ga >> 32) & 0x01FFFFFFu) | (2u << 30); // addr[56:32],type=2
  u32x8 g1;
  g1[0] = 1u << 16;                      // data_size=1 -> 2-byte elements
  g1[1] = (Kelems & 0xFFFFu) << 16;      // tensor_dim0[15:0]
  g1[2] = (Kelems >> 16) | ((rows & 0xFFFFu) << 16);  // dim0[31:16], dim1[15:0]
  g1[3] = (rows >> 16) | (64u << 16);    // dim1[31:16], tile_dim0 = 64
  g1[4] = 128u;                          // tile_dim1 = 128, tile_dim2 = 0
  g1[5] = Kelems;                        // tensor_dim0_stride[31:0]
  g1[6] = 0u;                            // stride[47:32], dim1_stride[15:0]
  g1[7] = 0u;
  asm volatile("tensor_load_to_lds %0, %1" :: "s"(g0), "s"(g1) : "memory");
}

// A/B fragment loader for V_WMMA_F32_16X16X32_F16 (ISA 7.12.2): lane<16 holds
// K = {b..b+7, b+16..b+23}, lane>=16 holds K = {b+8..b+15, b+24..b+31}.
__device__ __forceinline__ h16 frag_ab(const _Float16* rowp, int base1) {
  h8 lo = *(const h8*)(rowp + base1);
  h8 hi = *(const h8*)(rowp + base1 + 16);
  h16 r;
#pragma unroll
  for (int i = 0; i < 8; ++i) { r[i] = lo[i]; r[i + 8] = hi[i]; }
  return r;
}

__device__ __forceinline__ f8 wmma_f16(h16 a, h16 b, f8 c) {
  return __builtin_amdgcn_wmma_f32_16x16x32_f16(false, a, false, b, (short)0, c,
                                                false, false);
}

__global__ void cvt_f32_to_f16(const float* __restrict__ in,
                               _Float16* __restrict__ out, int n) {
  int i = blockIdx.x * blockDim.x + threadIdx.x;
  if (i < n) out[i] = (_Float16)in[i];
}

// C[M,N] = A[M,K]*W[N,K]^T + bias. 128x128 block tile, K-slab 64, TDM-staged
// double-buffered LDS. mode 1: f32 store. mode 0: QKV scatter epilogue
// (Q,K as [B,H,S,d] f16; V transposed as [B,H,d,S] f16).
__global__ __launch_bounds__(256)
void gemm_bt_wmma(const _Float16* __restrict__ A,
                  const _Float16* __restrict__ W,
                  const float* __restrict__ bias,
                  float* __restrict__ outF,
                  _Float16* __restrict__ qb,
                  _Float16* __restrict__ kb,
                  _Float16* __restrict__ vt,
                  int M, int N, int K, int mode) {
  __shared__ _Float16 sm[2][2][128 * 64];   // [buf][A|B][tile]  = 64 KB
  const int tid  = threadIdx.x;
  const int lane = tid & 31;
  const int wid  = tid >> 5;
  const int bm   = blockIdx.y * 128;
  const int bn   = blockIdx.x * 128;
  const int wm   = (wid >> 2) * 64;
  const int wn   = (wid & 3) * 32;
  const int l15  = lane & 15;
  const int b1   = (lane >> 4) * 8;

  f8 acc[4][2] = {};

  const _Float16* gA = A + (size_t)bm * K;   // block's A tile row 0
  const _Float16* gW = W + (size_t)bn * K;   // block's B tile row 0
  const uint32_t laA0 = lds_off(&sm[0][0][0]);
  const uint32_t laB0 = lds_off(&sm[0][1][0]);
  const uint32_t bufstep = (uint32_t)sizeof(_Float16) * 2 * 128 * 64;

  // Wave 0 drives the Tensor Data Mover. readfirstlane makes the predicate
  // scalar, forcing a wave-uniform s_cbranch: TDM ignores EXEC, so it must
  // never sit inside a saveexec-masked region with EXEC=0 waves on it.
  const bool tdm_wave = (__builtin_amdgcn_readfirstlane(wid) == 0);
  if (tdm_wave) {
    tdm_load_tile64(gA, laA0, (uint32_t)K, (uint32_t)M);
    tdm_load_tile64(gW, laB0, (uint32_t)K, (uint32_t)N);
  }
  int nb = 0;
  for (int k0 = 0; k0 < K; k0 += 64) {
    const bool more = (k0 + 64 < K);
    if (tdm_wave) {
      if (more) {
        const uint32_t o = (nb ^ 1) ? bufstep : 0u;
        tdm_load_tile64(gA + k0 + 64, laA0 + o, (uint32_t)K, (uint32_t)M);
        tdm_load_tile64(gW + k0 + 64, laB0 + o, (uint32_t)K, (uint32_t)N);
        __builtin_amdgcn_s_wait_tensorcnt(2);  // retire previous stage's pair
      } else {
        __builtin_amdgcn_s_wait_tensorcnt(0);
      }
    }
    __syncthreads();                           // consumers gate on barrier
    const _Float16* Ap = &sm[nb][0][0];
    const _Float16* Bp = &sm[nb][1][0];
#pragma unroll
    for (int ks = 0; ks < 2; ++ks) {
      const int kk = ks * 32 + b1;
      h16 bf0 = frag_ab(Bp + (wn + l15) * 64, kk);
      h16 bf1 = frag_ab(Bp + (wn + 16 + l15) * 64, kk);
#pragma unroll
      for (int i = 0; i < 4; ++i) {
        h16 af = frag_ab(Ap + (wm + i * 16 + l15) * 64, kk);
        acc[i][0] = wmma_f16(af, bf0, acc[i][0]);
        acc[i][1] = wmma_f16(af, bf1, acc[i][1]);
      }
    }
    __syncthreads();          // all reads of buf done before it is re-filled
    nb ^= 1;
  }

  // Epilogue. C layout: elem e -> row = e + 8*(lane>>4), col = lane&15.
#pragma unroll
  for (int i = 0; i < 4; ++i)
#pragma unroll
    for (int j = 0; j < 2; ++j)
#pragma unroll
      for (int e = 0; e < 8; ++e) {
        int gm = bm + wm + i * 16 + e + ((lane >> 4) << 3);
        int gn = bn + wn + j * 16 + l15;
        float v = acc[i][j][e] + bias[gn];
        if (mode == 1) {
          outF[(size_t)gm * N + gn] = v;
        } else {
          int bb = gm >> 11, s = gm & (S_ - 1);
          int hh = gn / 192, r = gn - hh * 192;
          int d = r & 63, which = r >> 6;     // 0=Q 1=K 2=V
          size_t bhid = (size_t)(bb * H_ + hh);
          _Float16 hv = (_Float16)v;
          if (which == 0)      qb[(bhid * S_ + s) * D_ + d] = hv;
          else if (which == 1) kb[(bhid * S_ + s) * D_ + d] = hv;
          else                 vt[(bhid * D_ + d) * S_ + s] = hv;
        }
      }
}

// Fused flash attention: 4 waves x 16 query rows per block; K/V slabs (32 keys)
// shared via double-buffered async LDS staging; online softmax; P routed
// through per-wave LDS tile to convert C-layout -> A-fragment layout.
__global__ __launch_bounds__(128)
void attn_wmma(const _Float16* __restrict__ qbuf,
               const _Float16* __restrict__ kbuf,
               const _Float16* __restrict__ vtb,
               _Float16* __restrict__ vals) {
  __shared__ _Float16 sK[2][32 * 64];   // keys x d        (4 KB each)
  __shared__ _Float16 sV[2][64 * 32];   // d x keys (V^T)  (4 KB each)
  __shared__ _Float16 Ps[4][16 * 32];   // per-wave P tile
  const int tid  = threadIdx.x;
  const int lane = tid & 31;
  const int wid  = tid >> 5;
  const int l15  = lane & 15;
  const int b1   = (lane >> 4) * 8;
  const int bh   = blockIdx.x >> 5;            // S_/64 = 32 q-tiles per (b,h)
  const int q0   = ((blockIdx.x & 31) << 6) + wid * 16;
  const int bb   = bh >> 4, hh = bh & 15;
  const _Float16* qp = qbuf + (size_t)bh * S_ * D_;
  const _Float16* kp = kbuf + (size_t)bh * S_ * D_;
  const _Float16* vp = vtb + (size_t)bh * D_ * S_;

  // Staging slices: K slab row=tid>>2, 16 f16 at (tid&3)*16; V^T slab
  // row=tid>>1, 16 f16 at (tid&1)*16. 4 async ops / thread / slab.
  const int kr = tid >> 2, kc = (tid & 3) * 16;
  const int vr = tid >> 1, vc = (tid & 1) * 16;
  const _Float16* gK = kp + (size_t)kr * D_ + kc;
  const _Float16* gV = vp + (size_t)vr * S_ + vc;
  const uint32_t laK = lds_off(&sK[0][kr * 64 + kc]);
  const uint32_t laV = lds_off(&sV[0][vr * 32 + vc]);
  const uint32_t stK = (uint32_t)sizeof(_Float16) * 32 * 64;
  const uint32_t stV = (uint32_t)sizeof(_Float16) * 64 * 32;

#define ATTN_ISSUE(buf, kt)                                                    \
  do {                                                                         \
    uint32_t _ok = (buf) ? stK : 0u, _ov = (buf) ? stV : 0u;                   \
    async_cp_b128(laK + _ok,      gK + (size_t)(kt) * D_);                     \
    async_cp_b128(laK + _ok + 16, gK + (size_t)(kt) * D_ + 8);                 \
    async_cp_b128(laV + _ov,      gV + (kt));                                  \
    async_cp_b128(laV + _ov + 16, gV + (kt) + 8);                              \
  } while (0)

  const h16 qa0 = frag_ab(qp + (size_t)(q0 + l15) * D_, b1);
  const h16 qa1 = frag_ab(qp + (size_t)(q0 + l15) * D_, 32 + b1);

  f8 acc[4] = {};
  float mrun[8], lrun[8];
#pragma unroll
  for (int i = 0; i < 8; ++i) { mrun[i] = -3.0e38f; lrun[i] = 0.f; }
  const float sc = 0.125f;                     // 1/sqrt(64)

  ATTN_ISSUE(0, 0);
  int nb = 0;
  for (int kt = 0; kt < S_; kt += 32) {
    if (kt + 32 < S_) {
      ATTN_ISSUE(nb ^ 1, kt + 32);
      ASYNC_WAIT(4);
    } else {
      ASYNC_WAIT(0);
    }
    __syncthreads();
    const _Float16* Kp = &sK[nb][0];
    const _Float16* Vp = &sV[nb][0];

    f8 c0 = {}, c1 = {};
    {
      h16 kf = frag_ab(Kp + l15 * 64, b1);
      c0 = wmma_f16(qa0, kf, c0);
      kf = frag_ab(Kp + l15 * 64, 32 + b1);
      c0 = wmma_f16(qa1, kf, c0);
      kf = frag_ab(Kp + (16 + l15) * 64, b1);
      c1 = wmma_f16(qa0, kf, c1);
      kf = frag_ab(Kp + (16 + l15) * 64, 32 + b1);
      c1 = wmma_f16(qa1, kf, c1);
    }
    float rescale[8];
#pragma unroll
    for (int i = 0; i < 8; ++i) {
      float x0 = c0[i] * sc, x1 = c1[i] * sc;
      float mx = fmaxf(x0, x1);                // row reduce inside 16-lane group
      mx = fmaxf(mx, __shfl_xor(mx, 1, 32));
      mx = fmaxf(mx, __shfl_xor(mx, 2, 32));
      mx = fmaxf(mx, __shfl_xor(mx, 4, 32));
      mx = fmaxf(mx, __shfl_xor(mx, 8, 32));
      float mnew = fmaxf(mrun[i], mx);
      float p0 = __expf(x0 - mnew), p1 = __expf(x1 - mnew);
      float rs = p0 + p1;
      rs += __shfl_xor(rs, 1, 32);
      rs += __shfl_xor(rs, 2, 32);
      rs += __shfl_xor(rs, 4, 32);
      rs += __shfl_xor(rs, 8, 32);
      float sold = __expf(mrun[i] - mnew);
      mrun[i] = mnew;
      lrun[i] = lrun[i] * sold + rs;
      rescale[i] = sold;
      int row = i + ((lane >> 4) << 3);        // C-layout -> row-major LDS
      Ps[wid][row * 32 + l15]      = (_Float16)p0;
      Ps[wid][row * 32 + 16 + l15] = (_Float16)p1;
    }
#pragma unroll
    for (int t = 0; t < 4; ++t)
#pragma unroll
      for (int i = 0; i < 8; ++i) acc[t][i] *= rescale[i];

    // Same-wave DS ops are in-order (ISA 7.3) -> safe to reload as A-frag.
    h16 pa = frag_ab(&Ps[wid][l15 * 32], b1);
#pragma unroll
    for (int t = 0; t < 4; ++t) {
      h16 vf = frag_ab(Vp + (t * 16 + l15) * 32, b1);
      acc[t] = wmma_f16(pa, vf, acc[t]);
    }
    __syncthreads();          // reads of slab done before it is re-filled
    nb ^= 1;
  }
#undef ATTN_ISSUE

#pragma unroll
  for (int t = 0; t < 4; ++t)
#pragma unroll
    for (int i = 0; i < 8; ++i) {
      int s = q0 + i + ((lane >> 4) << 3);
      int e = hh * D_ + t * 16 + l15;
      vals[((size_t)(bb * S_ + s)) * E_ + e] = (_Float16)(acc[t][i] / lrun[i]);
    }
}

extern "C" void kernel_launch(void* const* d_in, const int* in_sizes, int n_in,
                              void* d_out, int out_size, void* d_ws, size_t ws_size,
                              hipStream_t stream) {
  const float* x     = (const float*)d_in[0];
  const float* qkv_w = (const float*)d_in[1];
  const float* qkv_b = (const float*)d_in[2];
  const float* o_w   = (const float*)d_in[3];
  const float* o_b   = (const float*)d_in[4];
  float* out = (float*)d_out;

  // Workspace carve-up (f16): x, W1, W2, Q, K, V^T, attention output (~88 MB).
  _Float16* xh    = (_Float16*)d_ws;
  _Float16* w1h   = xh    + (size_t)M_ * E_;
  _Float16* w2h   = w1h   + (size_t)K3_ * E_;
  _Float16* qbuf  = w2h   + (size_t)E_ * E_;
  _Float16* kbuf  = qbuf  + (size_t)M_ * E_;
  _Float16* vtb   = kbuf  + (size_t)M_ * E_;
  _Float16* valsh = vtb   + (size_t)M_ * E_;

  const int nx = M_ * E_;
  const int n1 = K3_ * E_;
  const int n2 = E_ * E_;
  cvt_f32_to_f16<<<dim3((nx + 255) / 256), dim3(256), 0, stream>>>(x, xh, nx);
  cvt_f32_to_f16<<<dim3((n1 + 255) / 256), dim3(256), 0, stream>>>(qkv_w, w1h, n1);
  cvt_f32_to_f16<<<dim3((n2 + 255) / 256), dim3(256), 0, stream>>>(o_w, w2h, n2);

  // QKV projection (8192 x 3072 x 1024), scatter per-head Q/K/V^T.
  gemm_bt_wmma<<<dim3(K3_ / 128, M_ / 128), dim3(256), 0, stream>>>(
      xh, w1h, qkv_b, nullptr, qbuf, kbuf, vtb, M_, K3_, E_, 0);

  // Fused flash attention: B*H*(S/64) = 2048 four-wave blocks.
  attn_wmma<<<dim3(B_ * H_ * (S_ / 64)), dim3(128), 0, stream>>>(qbuf, kbuf, vtb,
                                                                 valsh);

  // Output projection (8192 x 1024 x 1024) + bias, f32 result.
  gemm_bt_wmma<<<dim3(E_ / 128, M_ / 128), dim3(256), 0, stream>>>(
      valsh, w2h, o_b, out, nullptr, nullptr, nullptr, M_, E_, E_, 1);
}
